// Attention_89721866814121
// MI455X (gfx1250) — matrix operand order repaired and checked by reading, not compiled, not run on hardware
//
#include <hip/hip_runtime.h>

typedef unsigned short u16;
typedef __bf16 bf16x16 __attribute__((ext_vector_type(16)));
typedef float  floatx8 __attribute__((ext_vector_type(8)));
typedef unsigned int uintx8 __attribute__((ext_vector_type(8)));
typedef unsigned int uintx4 __attribute__((ext_vector_type(4)));

__device__ __forceinline__ u16 f2bf(float f) {
    unsigned int u = __float_as_uint(f);
    unsigned int r = (u + 0x7FFFu + ((u >> 16) & 1u)) >> 16;
    return (u16)r;
}

__device__ __forceinline__ floatx8 wmma_bf16(bf16x16 a, bf16x16 b, floatx8 c) {
    return __builtin_amdgcn_wmma_f32_16x16x32_bf16(
        false, a, false, b, (short)0, c, false, false);
}

// A-matrix 16x32 bf16 fragment: lane holds one row;
// VGPR0-7 = K[half*16 .. +15] (contiguous, matches load_frag_b).
__device__ __forceinline__ bf16x16 load_frag_a(const u16* base, int row, int ld,
                                               int k0, int half) {
    const u16* p = base + (size_t)row * ld + k0;
    uintx4 lo = *(const uintx4*)(p + half * 16);
    uintx4 hi = *(const uintx4*)(p + half * 16 + 8);
    uintx8 r;
    r[0] = lo[0]; r[1] = lo[1]; r[2] = lo[2]; r[3] = lo[3];
    r[4] = hi[0]; r[5] = hi[1]; r[6] = hi[2]; r[7] = hi[3];
    return __builtin_bit_cast(bf16x16, r);
}

// B-matrix 32x16 bf16 fragment from N-major storage base[n][k]:
// lane n%16 holds 16 contiguous K values starting at k0 + half*16.
__device__ __forceinline__ bf16x16 load_frag_b(const u16* base, int nrow, int ld,
                                               int k0, int half) {
    const u16* p = base + (size_t)nrow * ld + k0 + half * 16;
    uintx8 r = *(const uintx8*)p;
    return __builtin_bit_cast(bf16x16, r);
}

// Generic shared pointer -> 32-bit LDS byte offset (flat->LDS keeps addr[31:0]).
__device__ __forceinline__ unsigned lds_off(const void* p) {
    return (unsigned)(unsigned long long)(size_t)p;
}

// Async-copy one 128x64 bf16 tile (row stride `ld` u16) into LDS.
// 1024 chunks of 16B; 4 per thread; 4 ASYNCcnt increments per wave.
__device__ __forceinline__ void async_tile_128x64(unsigned ldst, const u16* gsrc,
                                                  int row0, int ld, int k0, int tid) {
    #pragma unroll
    for (int i = 0; i < 4; ++i) {
        int idx = tid + i * 256;          // 0..1023
        int r = idx >> 3;                 // 8 x 16B chunks per row
        int c = (idx & 7) * 8;            // u16 units
        unsigned lo = ldst + (unsigned)(r * 128 + c * 2);
        unsigned long long ga =
            (unsigned long long)(size_t)(gsrc + (size_t)(row0 + r) * ld + k0 + c);
        asm volatile("global_load_async_to_lds_b128 %0, %1, off"
                     :: "v"(lo), "v"(ga) : "memory");
    }
}

// ---------------------------------------------------------------------------
// Kernel 0: one-shot conversions.
//  x (8192x1024 f32)       -> xbf   [m][k]  bf16
//  W_qkv (1024x3072 f32)   -> wqkvT [n][k]  bf16 (N-major)
//  W_proj (1024x1024 f32)  -> wprojT[n][k]  bf16 (N-major)
// ---------------------------------------------------------------------------
__global__ __launch_bounds__(256) void convert_kernel(
    const float* __restrict__ x, const float* __restrict__ Wqkv,
    const float* __restrict__ Wproj, u16* __restrict__ xbf,
    u16* __restrict__ wqkvT, u16* __restrict__ wprojT) {
    const size_t NX = 8192ull * 1024, NQ = 1024ull * 3072, NP = 1024ull * 1024;
    size_t i = (size_t)blockIdx.x * 256 + threadIdx.x;
    if (i < NX) {
        xbf[i] = f2bf(x[i]);
    } else if (i < NX + NQ) {
        size_t j = i - NX;
        size_t kk = j / 3072, n = j % 3072;
        wqkvT[n * 1024 + kk] = f2bf(Wqkv[j]);
    } else if (i < NX + NQ + NP) {
        size_t j = i - NX - NQ;
        size_t kk = j >> 10, n = j & 1023;
        wprojT[n * 1024 + kk] = f2bf(Wproj[j]);
    }
}

// ---------------------------------------------------------------------------
// Kernel 1: qkv = x @ W_qkv + b_qkv  (M=8192, N=3072, K=1024), bf16 WMMA.
// Double-buffered async LDS staging. Epilogue scatters bf16:
// q (scaled 1/8), k -> [B*H][S][64];  v -> [B*H][64][S] (transposed).
// ---------------------------------------------------------------------------
__global__ __launch_bounds__(256) void qkv_gemm_kernel(
    const u16* __restrict__ xbf, const u16* __restrict__ wT,
    const float* __restrict__ bias,
    u16* __restrict__ q, u16* __restrict__ k, u16* __restrict__ vT) {
    __shared__ __align__(16) u16 lds_a[2][128 * 64];
    __shared__ __align__(16) u16 lds_b[2][128 * 64];

    const int bm = blockIdx.x * 128;
    const int bn = blockIdx.y * 128;
    const int tid  = threadIdx.x;
    const int wave = tid >> 5, lane = tid & 31;
    const int half = lane >> 4, l16 = lane & 15;
    const int wm = (wave & 1) * 64;
    const int wn = (wave >> 1) * 32;

    floatx8 acc[4][2] = {};

    async_tile_128x64(lds_off(lds_a[0]), xbf, bm, 1024, 0, tid);
    async_tile_128x64(lds_off(lds_b[0]), wT,  bn, 1024, 0, tid);

    for (int k0 = 0; k0 < 1024; k0 += 64) {
        const int buf = (k0 >> 6) & 1;
        const bool pref = (k0 + 64) < 1024;
        if (pref) {
            async_tile_128x64(lds_off(lds_a[buf ^ 1]), xbf, bm, 1024, k0 + 64, tid);
            async_tile_128x64(lds_off(lds_b[buf ^ 1]), wT,  bn, 1024, k0 + 64, tid);
            asm volatile("s_wait_asynccnt 8" ::: "memory");   // current tile done
        } else {
            asm volatile("s_wait_asynccnt 0" ::: "memory");
        }
        __syncthreads();

        #pragma unroll
        for (int s = 0; s < 2; ++s) {
            bf16x16 bfrag[2];
            #pragma unroll
            for (int nt = 0; nt < 2; ++nt)
                bfrag[nt] = load_frag_b(lds_b[buf], wn + nt * 16 + l16, 64, s * 32, half);
            #pragma unroll
            for (int mt = 0; mt < 4; ++mt) {
                bf16x16 afrag = load_frag_a(lds_a[buf], wm + mt * 16 + l16, 64, s * 32, half);
                #pragma unroll
                for (int nt = 0; nt < 2; ++nt)
                    acc[mt][nt] = wmma_bf16(afrag, bfrag[nt], acc[mt][nt]);
            }
        }
        __syncthreads();
    }

    // Epilogue scatter (C layout: VGPR r -> row r + 8*half, col lane%16)
    #pragma unroll
    for (int mt = 0; mt < 4; ++mt)
        #pragma unroll
        for (int nt = 0; nt < 2; ++nt)
            #pragma unroll
            for (int r = 0; r < 8; ++r) {
                int m = bm + wm + mt * 16 + r + half * 8;
                int n = bn + wn + nt * 16 + l16;
                float val = acc[mt][nt][r] + bias[n];
                int t = n >> 10;            // 0:q 1:k 2:v
                int rem = n & 1023;
                int hh = rem >> 6, e = rem & 63;
                int b = m >> 10, s = m & 1023;
                size_t bh = (size_t)b * 16 + hh;
                if (t == 0)       q[(bh * 1024 + s) * 64 + e] = f2bf(val * 0.125f);
                else if (t == 1)  k[(bh * 1024 + s) * 64 + e] = f2bf(val);
                else              vT[(bh * 64 + e) * 1024 + s] = f2bf(val);
            }
}

// ---------------------------------------------------------------------------
// Kernel 2: flash attention per (b,h): out = softmax(q k^T) v
// grid (S/128, B*H); 8 waves, 16 query rows per wave; key tiles of 64.
// ---------------------------------------------------------------------------
__global__ __launch_bounds__(256) void flash_attn_kernel(
    const u16* __restrict__ q, const u16* __restrict__ k,
    const u16* __restrict__ vT, u16* __restrict__ attn) {
    __shared__ __align__(16) u16 lds_p[8][16 * 64];  // per-wave P tile (16x64 bf16)

    const int S = 1024, HD = 64;
    const int bh = blockIdx.y;
    const int tid  = threadIdx.x;
    const int wave = tid >> 5, lane = tid & 31;
    const int half = lane >> 4, l16 = lane & 15;
    const int qrow0 = blockIdx.x * 128 + wave * 16;

    const u16* qp = q  + (size_t)bh * S * HD;
    const u16* kp = k  + (size_t)bh * S * HD;
    const u16* vp = vT + (size_t)bh * HD * S;

    bf16x16 qfrag[2];
    qfrag[0] = load_frag_a(qp, qrow0 + l16, HD, 0, half);
    qfrag[1] = load_frag_a(qp, qrow0 + l16, HD, 32, half);

    floatx8 o[4] = {};
    float mrow[8], lrow[8];
    #pragma unroll
    for (int r = 0; r < 8; ++r) { mrow[r] = -1e30f; lrow[r] = 0.0f; }

    u16* pl = lds_p[wave];

    for (int kt = 0; kt < S; kt += 64) {
        floatx8 sacc[4] = {};
        #pragma unroll
        for (int nt = 0; nt < 4; ++nt)
            #pragma unroll
            for (int ks = 0; ks < 2; ++ks) {
                bf16x16 bf = load_frag_b(kp, kt + nt * 16 + l16, HD, ks * 32, half);
                sacc[nt] = wmma_bf16(qfrag[ks], bf, sacc[nt]);
            }

        float alpha[8];
        #pragma unroll
        for (int r = 0; r < 8; ++r) {
            float m = fmaxf(fmaxf(sacc[0][r], sacc[1][r]),
                            fmaxf(sacc[2][r], sacc[3][r]));
            #pragma unroll
            for (int off = 1; off < 16; off <<= 1)
                m = fmaxf(m, __shfl_xor(m, off, 32));
            float mnew = fmaxf(mrow[r], m);
            alpha[r] = __expf(mrow[r] - mnew);
            float ssum = 0.0f;
            #pragma unroll
            for (int nt = 0; nt < 4; ++nt) {
                float p = __expf(sacc[nt][r] - mnew);
                sacc[nt][r] = p;
                ssum += p;
            }
            #pragma unroll
            for (int off = 1; off < 16; off <<= 1)
                ssum += __shfl_xor(ssum, off, 32);
            lrow[r] = lrow[r] * alpha[r] + ssum;
            mrow[r] = mnew;
        }

        // P (C layout) -> LDS [16][64], reload in A layout (same-wave round trip)
        #pragma unroll
        for (int nt = 0; nt < 4; ++nt)
            #pragma unroll
            for (int r = 0; r < 8; ++r)
                pl[(r + 8 * half) * 64 + nt * 16 + l16] = f2bf(sacc[nt][r]);

        #pragma unroll
        for (int nt = 0; nt < 4; ++nt)
            #pragma unroll
            for (int r = 0; r < 8; ++r)
                o[nt][r] *= alpha[r];

        #pragma unroll
        for (int ks = 0; ks < 2; ++ks) {
            bf16x16 pfrag = load_frag_a(pl, l16, 64, ks * 32, half);
            #pragma unroll
            for (int nt = 0; nt < 4; ++nt) {
                bf16x16 vf = load_frag_b(vp, nt * 16 + l16, S, kt + ks * 32, half);
                o[nt] = wmma_bf16(pfrag, vf, o[nt]);
            }
        }
    }

    const int b = bh >> 4, hh = bh & 15;
    #pragma unroll
    for (int r = 0; r < 8; ++r) {
        float inv = 1.0f / lrow[r];
        int srow = qrow0 + r + 8 * half;
        size_t orow = ((size_t)b * S + srow) * 1024 + hh * 64;
        #pragma unroll
        for (int nt = 0; nt < 4; ++nt)
            attn[orow + nt * 16 + l16] = f2bf(o[nt][r] * inv);
    }
}

// ---------------------------------------------------------------------------
// Kernel 3: out = attn @ W_proj + b_proj  (M=8192, N=1024, K=1024), fp32 out.
// Same async double-buffered structure as kernel 1.
// ---------------------------------------------------------------------------
__global__ __launch_bounds__(256) void proj_gemm_kernel(
    const u16* __restrict__ attn, const u16* __restrict__ wT,
    const float* __restrict__ bias, float* __restrict__ out) {
    __shared__ __align__(16) u16 lds_a[2][128 * 64];
    __shared__ __align__(16) u16 lds_b[2][128 * 64];

    const int D = 1024;
    const int bm = blockIdx.x * 128;
    const int bn = blockIdx.y * 128;
    const int tid  = threadIdx.x;
    const int wave = tid >> 5, lane = tid & 31;
    const int half = lane >> 4, l16 = lane & 15;
    const int wm = (wave & 1) * 64;
    const int wn = (wave >> 1) * 32;

    floatx8 acc[4][2] = {};

    async_tile_128x64(lds_off(lds_a[0]), attn, bm, D, 0, tid);
    async_tile_128x64(lds_off(lds_b[0]), wT,   bn, D, 0, tid);

    for (int k0 = 0; k0 < D; k0 += 64) {
        const int buf = (k0 >> 6) & 1;
        const bool pref = (k0 + 64) < D;
        if (pref) {
            async_tile_128x64(lds_off(lds_a[buf ^ 1]), attn, bm, D, k0 + 64, tid);
            async_tile_128x64(lds_off(lds_b[buf ^ 1]), wT,   bn, D, k0 + 64, tid);
            asm volatile("s_wait_asynccnt 8" ::: "memory");
        } else {
            asm volatile("s_wait_asynccnt 0" ::: "memory");
        }
        __syncthreads();

        #pragma unroll
        for (int s = 0; s < 2; ++s) {
            bf16x16 bfrag[2];
            #pragma unroll
            for (int nt = 0; nt < 2; ++nt)
                bfrag[nt] = load_frag_b(lds_b[buf], wn + nt * 16 + l16, 64, s * 32, half);
            #pragma unroll
            for (int mt = 0; mt < 4; ++mt) {
                bf16x16 afrag = load_frag_a(lds_a[buf], wm + mt * 16 + l16, 64, s * 32, half);
                #pragma unroll
                for (int nt = 0; nt < 2; ++nt)
                    acc[mt][nt] = wmma_bf16(afrag, bfrag[nt], acc[mt][nt]);
            }
        }
        __syncthreads();
    }

    #pragma unroll
    for (int mt = 0; mt < 4; ++mt)
        #pragma unroll
        for (int nt = 0; nt < 2; ++nt)
            #pragma unroll
            for (int r = 0; r < 8; ++r) {
                int m = bm + wm + mt * 16 + r + half * 8;
                int n = bn + wn + nt * 16 + l16;
                out[(size_t)m * D + n] = acc[mt][nt][r] + bias[n];
            }
}

// ---------------------------------------------------------------------------
extern "C" void kernel_launch(void* const* d_in, const int* in_sizes, int n_in,
                              void* d_out, int out_size, void* d_ws, size_t ws_size,
                              hipStream_t stream) {
    const float* x      = (const float*)d_in[0];
    const float* W_qkv  = (const float*)d_in[1];
    const float* b_qkv  = (const float*)d_in[2];
    const float* W_proj = (const float*)d_in[3];
    const float* b_proj = (const float*)d_in[4];
    float* out = (float*)d_out;

    const size_t per = 128ull * 1024 * 64;          // 8,388,608 u16 (16 MB)
    u16* q      = (u16*)d_ws;
    u16* k      = q + per;
    u16* vT     = k + per;
    u16* attn   = vT + per;
    u16* xbf    = attn + per;                        // 8192*1024
    u16* wqkvT  = xbf + 8192ull * 1024;              // 3072*1024
    u16* wprojT = wqkvT + 3072ull * 1024;            // 1024*1024  (total 88 MB)

    const size_t ntot = 8192ull * 1024 + 1024ull * 3072 + 1024ull * 1024;
    convert_kernel<<<dim3((unsigned)(ntot / 256)), 256, 0, stream>>>(
        x, W_qkv, W_proj, xbf, wqkvT, wprojT);
    qkv_gemm_kernel<<<dim3(64, 24), 256, 0, stream>>>(xbf, wqkvT, b_qkv, q, k, vT);
    flash_attn_kernel<<<dim3(8, 128), 256, 0, stream>>>(q, k, vT, attn);
    proj_gemm_kernel<<<dim3(64, 8), 256, 0, stream>>>(attn, wqkvT /*unused slot*/ == wqkvT ? wprojT : wprojT, b_proj, out);
}
